// MultiHeadAttention_26036091748827
// MI455X (gfx1250) — matrix operand order repaired and checked by reading, not compiled, not run on hardware
//
#include <hip/hip_runtime.h>

// ---------------- types ----------------
typedef __attribute__((ext_vector_type(16))) _Float16 v16h;
typedef __attribute__((ext_vector_type(8)))  float    v8f;
typedef int v4i32 __attribute__((vector_size(16)));            // matches async builtin pointee
typedef unsigned int u32x4 __attribute__((ext_vector_type(4)));
typedef int i32x8 __attribute__((ext_vector_type(8)));
typedef int i32x4 __attribute__((ext_vector_type(4)));

#define D_MODEL 1024
#define NUM_HEADS 16
#define DEPTH 64
#define BATCH 4
#define SEQ 2048
#define MROWS (BATCH * SEQ)   // 8192

union FragU { int i[8]; v16h v; };

static __device__ __forceinline__ v8f wmma_f16(v16h a, v16h b, v8f c) {
  return __builtin_amdgcn_wmma_f32_16x16x32_f16(false, a, false, b, (short)0, c, false, false);
}

// 16-byte global -> LDS async copy (ASYNCcnt; bypasses VGPRs).
static __device__ __forceinline__ void copy16(_Float16* lds, const _Float16* g) {
  __builtin_amdgcn_global_load_async_to_lds_b128(
      (__attribute__((address_space(1))) v4i32*)(uintptr_t)g,
      (__attribute__((address_space(3))) v4i32*)(unsigned)(uintptr_t)lds,
      0, 0);
}

static __device__ __forceinline__ void wait_async() {
  __builtin_amdgcn_s_wait_asynccnt(0);
}

static __device__ __forceinline__ void wait_tensor() {
  __builtin_amdgcn_s_wait_tensorcnt(0);
}

// TDM: DMA a 2D tile (tile_x elems of 2B, tile_y rows, row stride in elems)
// from global memory into LDS (contiguous row-major). D# per CDNA5 ISA 8.3/8.4.
static __device__ __forceinline__ void tdm_load_tile_2d(
    const void* gsrc, unsigned lds_bytes,
    unsigned tile_x, unsigned tile_y, unsigned row_stride_elems) {
  const unsigned long long ga = (unsigned long long)(uintptr_t)gsrc;
  u32x4 g0;
  g0.x = 1u;                                                  // count=1 (valid), user mode
  g0.y = lds_bytes;                                           // lds_addr (bytes)
  g0.z = (unsigned)ga;                                        // global_addr[31:0]
  g0.w = (unsigned)((ga >> 32) & 0x01FFFFFFu) | (2u << 30);   // global_addr[56:32] | type=2
  i32x8 g1;
  g1[0] = (int)(1u << 16);                                    // workgroup_mask=0, data_size=1 (2B)
  g1[1] = (int)((tile_x & 0xFFFFu) << 16);                    // tensor_dim0[15:0] (= tile_x)
  g1[2] = (int)(((tile_x >> 16) & 0xFFFFu) | ((tile_y & 0xFFFFu) << 16)); // dim0 hi | tensor_dim1 lo
  g1[3] = (int)(((tile_y >> 16) & 0xFFFFu) | ((tile_x & 0xFFFFu) << 16)); // dim1 hi | tile_dim0
  g1[4] = (int)(tile_y & 0xFFFFu);                            // tile_dim1 | tile_dim2=0
  g1[5] = (int)row_stride_elems;                              // tensor_dim0_stride[31:0]
  g1[6] = 0;                                                  // stride hi | dim1_stride lo
  g1[7] = 0;
  i32x4 z4 = {0, 0, 0, 0};
  i32x8 z8 = {0, 0, 0, 0, 0, 0, 0, 0};
  __builtin_amdgcn_tensor_load_to_lds(g0, g1, z4, z4, z8, 0);
}

// A-fragment (16x32, f16) from LDS, row-major tile with row stride `ld` halfwords.
static __device__ __forceinline__ v16h load_a_frag(const _Float16* lds, int row0, int k0, int ld) {
  const int lane = threadIdx.x & 31;
  const int m = lane & 15, kg = lane >> 4;
  const _Float16* p = lds + (row0 + m) * ld + k0 + kg * 16;
  FragU u;
#pragma unroll
  for (int j = 0; j < 8; ++j) {
    const int kb = 2 * j;
    u.i[j] = *(const int*)(p + kb);
  }
  return u.v;
}

// B-fragment (32x16 KxN, f16) from LDS stored N-major: lds[n*ld + k].
static __device__ __forceinline__ v16h load_b_frag(const _Float16* lds, int n0, int k0, int ld) {
  const int lane = threadIdx.x & 31;
  const int n = lane & 15, kh = (lane >> 4) * 16;
  const _Float16* p = lds + (n0 + n) * ld + k0 + kh;
  FragU u;
#pragma unroll
  for (int j = 0; j < 8; ++j) u.i[j] = *(const int*)(p + 2 * j);
  return u.v;
}

// ---------------- cast kernels ----------------
__global__ void cast_f32_to_f16(const float* __restrict__ in, _Float16* __restrict__ out, int n) {
  int i = blockIdx.x * blockDim.x + threadIdx.x;
  if (i < n) out[i] = (_Float16)in[i];
}

// Wt[n*K + k] = (f16) W[k*N + n]   (K = N = 1024)
__global__ void cast_transpose_f32_to_f16(const float* __restrict__ W, _Float16* __restrict__ Wt, int n) {
  int i = blockIdx.x * blockDim.x + threadIdx.x;
  if (i < n) {
    int nn = i >> 10;
    int kk = i & 1023;
    Wt[i] = (_Float16)W[(size_t)kk * D_MODEL + nn];
  }
}

// ---------------- tiled WMMA GEMM (TDM tile staging, 4x4 frags/wave) ----------------
// C[M,N] = A[M,K] * Bt[N,K]^T + bias[N].  Block tile 256x128, BK=64, double-buffered TDM.
// 8 waves in 4(M) x 2(N); each wave computes a 64x64 output block = 16 accum fragments.
template <bool OUT_F32>
__global__ __launch_bounds__(256)
void gemm_wmma(const _Float16* __restrict__ A, const _Float16* __restrict__ Bt,
               const float* __restrict__ bias, void* __restrict__ Cout,
               int M, int N, int K) {
  __shared__ __align__(16) _Float16 As[2][256 * 64];   // 64 KB
  __shared__ __align__(16) _Float16 Bs[2][128 * 64];   // 32 KB

  const int tid = threadIdx.x;
  const int lane = tid & 31;
  const int wave = tid >> 5;
  const int wm = wave & 3;          // 64-row strip
  const int wn = wave >> 2;         // 64-col strip
  const int m0 = blockIdx.y * 256;
  const int n0 = blockIdx.x * 128;

  v8f acc[4][4] = {};

  auto load_tiles = [&](int buf, int k0) {
    if (wave == 0) {   // TDM ignores EXEC; one DMA per tile, issued by wave 0
      tdm_load_tile_2d(A  + (size_t)m0 * K + k0, (unsigned)(uintptr_t)&As[buf][0], 64, 256, (unsigned)K);
      tdm_load_tile_2d(Bt + (size_t)n0 * K + k0, (unsigned)(uintptr_t)&Bs[buf][0], 64, 128, (unsigned)K);
    }
  };

  load_tiles(0, 0);
  wait_tensor();          // wave0 waits for DMA; others have TENSORcnt==0
  __syncthreads();

  int cur = 0;
  for (int k0 = 0; k0 < K; k0 += 64) {
    if (k0 + 64 < K) load_tiles(cur ^ 1, k0 + 64);   // DMA overlaps WMMAs below

#pragma unroll
    for (int kk = 0; kk < 64; kk += 32) {
      v16h af[4], bf[4];
#pragma unroll
      for (int i = 0; i < 4; ++i) af[i] = load_a_frag(&As[cur][0], wm * 64 + i * 16, kk, 64);
#pragma unroll
      for (int f = 0; f < 4; ++f) bf[f] = load_b_frag(&Bs[cur][0], wn * 64 + f * 16, kk, 64);
#pragma unroll
      for (int i = 0; i < 4; ++i)
#pragma unroll
        for (int f = 0; f < 4; ++f) acc[i][f] = wmma_f16(af[i], bf[f], acc[i][f]);
    }

    wait_tensor();
    __syncthreads();
    cur ^= 1;
  }

  const int nlo = lane & 15;
  const int hi = lane >> 4;
#pragma unroll
  for (int i = 0; i < 4; ++i)
#pragma unroll
    for (int f = 0; f < 4; ++f) {
      const int col = n0 + wn * 64 + f * 16 + nlo;
      const float bv = bias[col];
#pragma unroll
      for (int r = 0; r < 8; ++r) {
        const int row = m0 + wm * 64 + i * 16 + r + 8 * hi;
        const float v = acc[i][f][r] + bv;
        if (OUT_F32) ((float*)Cout)[(size_t)row * N + col] = v;
        else         ((_Float16*)Cout)[(size_t)row * N + col] = (_Float16)v;
      }
    }
}

// ---------------- flash attention (async-LDS staging) ----------------
// grid: (SEQ/64, BATCH*NUM_HEADS), block: 128 threads (4 waves x 16 q-rows).
__global__ __launch_bounds__(128)
void mha_flash(const _Float16* __restrict__ Q, const _Float16* __restrict__ Km,
               const _Float16* __restrict__ V, _Float16* __restrict__ ctx) {
  __shared__ __align__(16) _Float16 Qs[64 * 64];
  __shared__ __align__(16) _Float16 Ks[2][64 * 64];
  __shared__ __align__(16) _Float16 Vraw[2][64 * 64];  // row-major [j][d]
  __shared__ __align__(16) _Float16 Vts[64 * 64];      // transposed [d][j]
  __shared__ __align__(16) _Float16 Ps[64 * 64];

  const int tid = threadIdx.x;
  const int lane = tid & 31;
  const int wave = tid >> 5;
  const int qb0 = blockIdx.x * 64;
  const int bh = blockIdx.y;
  const int b = bh >> 4, h = bh & 15;
  const size_t base = ((size_t)b * SEQ) * D_MODEL + h * DEPTH;

  auto load_kv = [&](int buf, int kb) {
#pragma unroll
    for (int q = 0; q < 4; ++q) {
      const int c = tid * 4 + q;          // 512 chunks per 64x64 tile
      const int row = c >> 3;
      const int seg = (c & 7) * 8;
      copy16(&Ks[buf][row * 64 + seg],   Km + base + (size_t)(kb + row) * D_MODEL + seg);
      copy16(&Vraw[buf][row * 64 + seg], V  + base + (size_t)(kb + row) * D_MODEL + seg);
    }
  };

  // Q tile + first K/V block (async)
#pragma unroll
  for (int q = 0; q < 4; ++q) {
    const int c = tid * 4 + q;
    const int row = c >> 3;
    const int seg = (c & 7) * 8;
    copy16(&Qs[row * 64 + seg], Q + base + (size_t)(qb0 + row) * D_MODEL + seg);
  }
  load_kv(0, 0);
  wait_async();
  __syncthreads();

  float m[8], l[8];
  v8f O[4] = {};
#pragma unroll
  for (int r = 0; r < 8; ++r) { m[r] = -3.0e38f; l[r] = 0.0f; }

  const int nlo = lane & 15;
  const int hi = lane >> 4;

  int cur = 0;
  for (int kb = 0; kb < SEQ; kb += 64) {
    if (kb + 64 < SEQ) load_kv(cur ^ 1, kb + 64);   // async prefetch next block

    // LDS->LDS transpose of V tile: Vts[d][j] = Vraw[j][d]
    {
      const int row = tid >> 1, cs = (tid & 1) * 32;
      const _Float16* src = &Vraw[cur][row * 64 + cs];
#pragma unroll
      for (int d = 0; d < 32; d += 2) {
        const int u = *(const int*)(src + d);
        _Float16 v2[2];
        *(int*)v2 = u;
        Vts[(cs + d) * 64 + row]     = v2[0];
        Vts[(cs + d + 1) * 64 + row] = v2[1];
      }
    }
    __syncthreads();   // Vts ready for all waves

    // ---- scores: S = Q * K^T ----
    v8f sacc[4] = {};
#pragma unroll
    for (int kk = 0; kk < 64; kk += 32) {
      v16h aq = load_a_frag(Qs, wave * 16, kk, 64);
#pragma unroll
      for (int f = 0; f < 4; ++f) {
        v16h bk = load_b_frag(&Ks[cur][0], f * 16, kk, 64);
        sacc[f] = wmma_f16(aq, bk, sacc[f]);
      }
    }

    // ---- online softmax (fp32) ----
#pragma unroll
    for (int f = 0; f < 4; ++f)
#pragma unroll
      for (int r = 0; r < 8; ++r) sacc[f][r] *= 0.125f;

    float mnew[8];
#pragma unroll
    for (int r = 0; r < 8; ++r)
      mnew[r] = fmaxf(fmaxf(sacc[0][r], sacc[1][r]), fmaxf(sacc[2][r], sacc[3][r]));
#pragma unroll
    for (int off = 1; off <= 8; off <<= 1)
#pragma unroll
      for (int r = 0; r < 8; ++r) mnew[r] = fmaxf(mnew[r], __shfl_xor(mnew[r], off, 32));

    float alpha[8];
#pragma unroll
    for (int r = 0; r < 8; ++r) {
      const float mo = m[r];
      const float mn = fmaxf(mo, mnew[r]);
      m[r] = mn;
      alpha[r] = __expf(mo - mn);
    }
#pragma unroll
    for (int f = 0; f < 4; ++f)
#pragma unroll
      for (int r = 0; r < 8; ++r) sacc[f][r] = __expf(sacc[f][r] - m[r]);

    float rs[8];
#pragma unroll
    for (int r = 0; r < 8; ++r)
      rs[r] = (sacc[0][r] + sacc[1][r]) + (sacc[2][r] + sacc[3][r]);
#pragma unroll
    for (int off = 1; off <= 8; off <<= 1)
#pragma unroll
      for (int r = 0; r < 8; ++r) rs[r] += __shfl_xor(rs[r], off, 32);
#pragma unroll
    for (int r = 0; r < 8; ++r) l[r] = l[r] * alpha[r] + rs[r];
#pragma unroll
    for (int f = 0; f < 4; ++f)
#pragma unroll
      for (int r = 0; r < 8; ++r) O[f][r] *= alpha[r];

    // ---- P -> f16 in LDS (per-wave rows, wave-local readback) ----
#pragma unroll
    for (int f = 0; f < 4; ++f)
#pragma unroll
      for (int r = 0; r < 8; ++r)
        Ps[(wave * 16 + r + 8 * hi) * 64 + f * 16 + nlo] = (_Float16)sacc[f][r];

    // ---- O += P * V ----
#pragma unroll
    for (int kk = 0; kk < 64; kk += 32) {
      v16h ap = load_a_frag(Ps, wave * 16, kk, 64);
#pragma unroll
      for (int f = 0; f < 4; ++f) {
        v16h bv = load_b_frag(Vts, f * 16, kk, 64);
        O[f] = wmma_f16(ap, bv, O[f]);
      }
    }

    wait_async();
    __syncthreads();
    cur ^= 1;
  }

  // normalize and store context
#pragma unroll
  for (int f = 0; f < 4; ++f)
#pragma unroll
    for (int r = 0; r < 8; ++r) {
      const float v = O[f][r] / l[r];
      const int qrow = qb0 + wave * 16 + r + 8 * hi;
      ctx[base + (size_t)qrow * D_MODEL + f * 16 + nlo] = (_Float16)v;
    }
}

// ---------------- launcher ----------------
extern "C" void kernel_launch(void* const* d_in, const int* in_sizes, int n_in,
                              void* d_out, int out_size, void* d_ws, size_t ws_size,
                              hipStream_t stream) {
  const float* x  = (const float*)d_in[0];
  const float* Wq = (const float*)d_in[1];
  const float* bq = (const float*)d_in[2];
  const float* Wk = (const float*)d_in[3];
  const float* bk = (const float*)d_in[4];
  const float* Wv = (const float*)d_in[5];
  const float* bv = (const float*)d_in[6];
  const float* Wo = (const float*)d_in[7];
  const float* bo = (const float*)d_in[8];

  _Float16* ws = (_Float16*)d_ws;
  const size_t XN = (size_t)MROWS * D_MODEL;
  const size_t WN = (size_t)D_MODEL * D_MODEL;
  _Float16* xb   = ws;
  _Float16* wqt  = xb + XN;
  _Float16* wkt  = wqt + WN;
  _Float16* wvt  = wkt + WN;
  _Float16* wot  = wvt + WN;
  _Float16* qb   = wot + WN;
  _Float16* kbuf = qb + XN;
  _Float16* vbuf = kbuf + XN;
  _Float16* ctx  = vbuf + XN;

  cast_f32_to_f16<<<(int)(XN / 256), 256, 0, stream>>>(x, xb, (int)XN);
  cast_transpose_f32_to_f16<<<(int)(WN / 256), 256, 0, stream>>>(Wq, wqt, (int)WN);
  cast_transpose_f32_to_f16<<<(int)(WN / 256), 256, 0, stream>>>(Wk, wkt, (int)WN);
  cast_transpose_f32_to_f16<<<(int)(WN / 256), 256, 0, stream>>>(Wv, wvt, (int)WN);
  cast_transpose_f32_to_f16<<<(int)(WN / 256), 256, 0, stream>>>(Wo, wot, (int)WN);

  dim3 ggrid(D_MODEL / 128, MROWS / 256);
  gemm_wmma<false><<<ggrid, 256, 0, stream>>>(xb, wqt, bq, qb,   MROWS, D_MODEL, D_MODEL);
  gemm_wmma<false><<<ggrid, 256, 0, stream>>>(xb, wkt, bk, kbuf, MROWS, D_MODEL, D_MODEL);
  gemm_wmma<false><<<ggrid, 256, 0, stream>>>(xb, wvt, bv, vbuf, MROWS, D_MODEL, D_MODEL);

  mha_flash<<<dim3(SEQ / 64, BATCH * NUM_HEADS), 128, 0, stream>>>(qb, kbuf, vbuf, ctx);

  gemm_wmma<true><<<ggrid, 256, 0, stream>>>(ctx, wot, bo, d_out, MROWS, D_MODEL, D_MODEL);
}
